// MapFeatureExtractor_30640296689740
// MI455X (gfx1250) — matrix operand order repaired and checked by reading, compiled, not run on hardware
//
#include <hip/hip_runtime.h>

typedef __attribute__((ext_vector_type(16))) _Float16 v16h;
typedef __attribute__((ext_vector_type(8)))  _Float16 v8h;
typedef __attribute__((ext_vector_type(8)))  float    v8f;

#define BATCH   4
#define NQ      4096
#define MS      8192
#define WAVES   8
#define THREADS (WAVES * 32)
#define SUPER   2048                 // scene points staged in LDS per pass
#define PER_WAVE (SUPER / WAVES)     // 256 m-values per wave per pass
#define CHUNK   32                   // K per WMMA

// log2(e)^2 : exp(-d) = exp2(-sqrt((dx^2+dy^2) * LOG2E_SQ))  (neg folds into v_exp src mod)
#define LOG2E_SQ 2.0813689810056077f

// out[b,n] = (sum_m exp(-||q_xy - p_xy||) * feat[m]) / (sum_m exp(-||q_xy - p_xy||))
__global__ __launch_bounds__(THREADS)
void MapFeatureExtractor_kernel(const float* __restrict__ q,    // [B, NQ, 3]
                                const float* __restrict__ pc,   // [B, MS, 3]
                                const float* __restrict__ ft,   // [B, MS, 1]
                                float* __restrict__ out)        // [B, NQ, 1]
{
    __shared__ float    s_x[SUPER];                                  // 8 KB
    __shared__ float    s_y[SUPER];                                  // 8 KB
    __shared__ _Float16 s_f[SUPER] __attribute__((aligned(32)));     // 4 KB
    __shared__ float    s_part[WAVES * 32 * 8];                      // 8 KB

    const int wg   = blockIdx.x;            // 0 .. B*(NQ/16)-1
    const int tile = wg % (NQ / 16);
    const int b    = wg / (NQ / 16);
    const int t    = threadIdx.x;
    const int wave = t >> 5;
    const int lane = t & 31;
    const int half = lane >> 4;             // K-half selector for A layout
    const int r    = lane & 15;             // A row (query row) / B column

    // Per-lane query point (first 2 coords), held in registers for whole kernel.
    const int   qi = (b * NQ + tile * 16 + r) * 3;
    const float qx = q[qi + 0];
    const float qy = q[qi + 1];

    // Branch-free B-column masks (hoisted out of all loops):
    // column 0 gets feats, column 1 gets ones, rest zero.
    const _Float16 m0 = (r == 0) ? (_Float16)1.0f : (_Float16)0.0f;
    const _Float16 m1 = (r == 1) ? (_Float16)1.0f : (_Float16)0.0f;
    v16h selF, sel1;
#pragma unroll
    for (int e = 0; e < 16; ++e) { selF[e] = m0; sel1[e] = m1; }

    v8f c = {};                             // f32 accumulator: D[:,0]=Σw*f, D[:,1]=Σw

    for (int s0 = 0; s0 < MS; s0 += SUPER) {
        __syncthreads();                    // protect LDS from previous pass readers
        // Cooperative stage of 2048 scene points into LDS (SoA, feat as f16).
        for (int i = t; i < SUPER; i += THREADS) {
            const int m = b * MS + s0 + i;
            s_x[i] = pc[m * 3 + 0];
            s_y[i] = pc[m * 3 + 1];
            s_f[i] = (_Float16)ft[m];
            if (s0 + SUPER < MS)            // prefetch next super-chunk into caches
                __builtin_prefetch(&pc[(m + SUPER) * 3], 0, 1);
        }
        __syncthreads();

        const int wbase = wave * PER_WAVE;
        for (int ch = 0; ch < PER_WAVE; ch += CHUNK) {
            const int kbase = wbase + ch;   // LDS index of this chunk's K=0

            // A (16x32 f16 weights): lane holds row r; element e -> K (matches B):
            // K = 16*(lane/16) + e  -> 16 consecutive m.
            v16h a;
#pragma unroll
            for (int e = 0; e < 16; ++e) {
                const int   K  = (half << 4) + e;
                const float dx = qx - s_x[kbase + K];
                const float dy = qy - s_y[kbase + K];
                const float s  = (dx * dx + dy * dy) * LOG2E_SQ;
                const float w  = __builtin_amdgcn_exp2f(-__builtin_amdgcn_sqrtf(s));
                a[e] = (_Float16)w;
            }

            // B (32x16 f16): lane holds column r; K = 16*half + e (16 consecutive f16
            // = 32 aligned bytes -> two ds_load_b128). Branch-free column select.
            const v8h* pf = (const v8h*)&s_f[kbase + (half << 4)];
            const v8h  f0 = pf[0];
            const v8h  f1 = pf[1];
            v16h fv;
#pragma unroll
            for (int e = 0; e < 8; ++e) { fv[e] = f0[e]; fv[e + 8] = f1[e]; }
            const v16h bm = fv * selF + sel1;   // 8x v_pk_fma_f16, no branches

            c = __builtin_amdgcn_wmma_f32_16x16x32_f16(
                    /*neg_a=*/false, a, /*neg_b=*/false, bm,
                    /*c_mod=*/(short)0, c, /*reuse_a=*/false, /*reuse_b=*/false);
        }
    }

    // Dump each wave's 16x16 C tile to LDS for a deterministic cross-wave reduce.
#pragma unroll
    for (int g = 0; g < 8; ++g)
        s_part[(wave * 32 + lane) * 8 + g] = c[g];
    __syncthreads();

    // C layout: lane l holds column N=l%16 of rows M = g + 8*(l/16).
    // Row M lives in lane (M/8)*16 + N at VGPR g = M%8.
    if (t < 16) {
        const int Mrow = t;
        const int ln0  = (Mrow >> 3) << 4;  // lane with N=0 for this row
        const int g    = Mrow & 7;
        float num = 0.f, den = 0.f;
#pragma unroll
        for (int w = 0; w < WAVES; ++w) {
            num += s_part[(w * 32 + ln0 + 0) * 8 + g];   // N=0: Σ w*f
            den += s_part[(w * 32 + ln0 + 1) * 8 + g];   // N=1: Σ w
        }
        out[b * NQ + tile * 16 + Mrow] = num / den;
    }
}

extern "C" void kernel_launch(void* const* d_in, const int* in_sizes, int n_in,
                              void* d_out, int out_size, void* d_ws, size_t ws_size,
                              hipStream_t stream) {
    const float* q  = (const float*)d_in[0];   // [4, 4096, 3]
    const float* pc = (const float*)d_in[1];   // [4, 8192, 3]
    const float* ft = (const float*)d_in[2];   // [4, 8192, 1]
    float* out = (float*)d_out;                // [4, 4096, 1]

    const int grid = BATCH * (NQ / 16);        // 1024 workgroups, 256 threads each
    MapFeatureExtractor_kernel<<<grid, THREADS, 0, stream>>>(q, pc, ft, out);
}
